// MMFLinear_25305947308549
// MI455X (gfx1250) — matrix operand order repaired and checked by reading, not compiled, not run on hardware
//
#include <hip/hip_runtime.h>

// ---------------------------------------------------------------------------
// Ternary linear: out[B,O] = scale * (x[B,I] @ W[O,I]^T) + bias[O]
// B = I = O = 4096. bf16 WMMA (v_wmma_f32_16x16x32_bf16): ternary weights are
// exact in bf16; both bf16 operands (64 MB) fit in the 192 MB L2.
// Tile staging via GLOBAL_LOAD_ASYNC_TO_LDS_B128 (ASYNCcnt) with literal
// immediate chunk offsets (IOFFSET applies to both LDS and global address).
// BK=64: 32 WMMAs per barrier interval.
// ---------------------------------------------------------------------------

typedef __attribute__((ext_vector_type(16))) __bf16  v16bf;
typedef __attribute__((ext_vector_type(8)))  float   v8f;
typedef __attribute__((ext_vector_type(4)))  float   v4f;
typedef __attribute__((ext_vector_type(4)))  unsigned int u32x4;
typedef __attribute__((ext_vector_type(4)))  int     v4i;

#define DIM_B 4096
#define DIM_I 4096
#define DIM_O 4096

#define BM 128      // block tile M
#define BN 128      // block tile N
#define BK 64       // K step (two WMMA K chunks)
#define LDK 72      // padded LDS row length in bf16 elements (144B, conflict-free)
#define NT (DIM_I / BK)

#if __has_builtin(__builtin_amdgcn_global_load_async_to_lds_b128)
#define HAVE_ASYNC_LDS 1
typedef __attribute__((address_space(1))) v4i gv4i;   // global int4
typedef __attribute__((address_space(3))) v4i lv4i;   // LDS int4
#else
#define HAVE_ASYNC_LDS 0
#endif

static __device__ __forceinline__ void wait_async0() {
#if __has_builtin(__builtin_amdgcn_s_wait_asynccnt)
    __builtin_amdgcn_s_wait_asynccnt(0);
#else
    asm volatile("s_wait_asynccnt 0x0" ::: "memory");
#endif
}

static __device__ __forceinline__ unsigned short f2bf(float f) {
    unsigned u = __builtin_bit_cast(unsigned, f);
    u += 0x7FFFu + ((u >> 16) & 1u);          // round-to-nearest-even
    return (unsigned short)(u >> 16);
}
static __device__ __forceinline__ unsigned pack2(float lo, float hi) {
    return (unsigned)f2bf(lo) | ((unsigned)f2bf(hi) << 16);
}

// ---------------------------------------------------------------------------
// f32 -> bf16 conversion pre-pass (8 elements / thread)
// ---------------------------------------------------------------------------
__global__ __launch_bounds__(256)
void cvt_f32_bf16_kernel(const float* __restrict__ src,
                         unsigned short* __restrict__ dst) {
    size_t i = ((size_t)blockIdx.x * blockDim.x + threadIdx.x) * 8;
    v4f f0 = *(const v4f*)(src + i);
    v4f f1 = *(const v4f*)(src + i + 4);
    u32x4 o;
    o.x = pack2(f0.x, f0.y);
    o.y = pack2(f0.z, f0.w);
    o.z = pack2(f1.x, f1.y);
    o.w = pack2(f1.z, f1.w);
    *(u32x4*)(dst + i) = o;
}

// ---------------------------------------------------------------------------
// Tiled bf16 WMMA GEMM with fused scale/bias epilogue.
// PRE=true : A/W already bf16 (from d_ws pre-pass) -> async-to-LDS staging.
// PRE=false: A/W are f32; convert inline while staging to LDS.
// ---------------------------------------------------------------------------
union Frag { u32x4 u[2]; v16bf v; };

template<bool PRE>
__global__ __launch_bounds__(128)
void ternary_gemm_kernel(const void* __restrict__ Ag,   // x  [B,I]
                         const void* __restrict__ Wg,   // W  [O,I]
                         const float* __restrict__ bias,
                         const float* __restrict__ scalep,
                         float* __restrict__ out) {
    __shared__ unsigned short sA[2][BM * LDK];
    __shared__ unsigned short sW[2][BN * LDK];

    const int tid  = threadIdx.x;          // 0..127
    const int lane = tid & 31;
    const int wave = tid >> 5;             // 0..3
    const int wm   = (wave >> 1) * 64;     // wave tile origin in block (M)
    const int wn   = (wave & 1) * 64;      // wave tile origin in block (N)

    const int m0 = blockIdx.y * BM;
    const int n0 = blockIdx.x * BN;

    const int l15 = lane & 15;
    const bool hi = lane >= 16;
    const int kbA = hi ? 16 : 0;           // A frag: contiguous K run start
    const int ksB = hi ? 16 : 0;           // B frag: contiguous K run start

    v8f acc[4][4] = {};

    auto load_tiles = [&](int kt, int buf) {
        const int k0 = kt * BK;
        unsigned short* dra = &sA[buf][tid * LDK];
        unsigned short* drw = &sW[buf][tid * LDK];
        if constexpr (PRE) {
            const unsigned short* ga = (const unsigned short*)Ag +
                                       (size_t)(m0 + tid) * DIM_I + k0;
            const unsigned short* gw = (const unsigned short*)Wg +
                                       (size_t)(n0 + tid) * DIM_I + k0;
#if HAVE_ASYNC_LDS
            // IOFFSET applies to both the LDS and global address -> one base
            // pointer per tile, 8 literal-offset async b128 loads (128B/row).
#define ACP(OFF)                                                              \
            __builtin_amdgcn_global_load_async_to_lds_b128(                   \
                (gv4i*)ga, (lv4i*)dra, (OFF), 0);                             \
            __builtin_amdgcn_global_load_async_to_lds_b128(                   \
                (gv4i*)gw, (lv4i*)drw, (OFF), 0)
            ACP(0); ACP(16); ACP(32); ACP(48);
            ACP(64); ACP(80); ACP(96); ACP(112);
#undef ACP
#else
#pragma unroll
            for (int j = 0; j < 8; ++j) {
                ((u32x4*)dra)[j] = ((const u32x4*)ga)[j];
                ((u32x4*)drw)[j] = ((const u32x4*)gw)[j];
            }
#endif
        } else {
            const float* ga = (const float*)Ag + (size_t)(m0 + tid) * DIM_I + k0;
            const float* gw = (const float*)Wg + (size_t)(n0 + tid) * DIM_I + k0;
#pragma unroll
            for (int j = 0; j < 8; ++j) {
                v4f f0 = *(const v4f*)(ga + j * 8);
                v4f f1 = *(const v4f*)(ga + j * 8 + 4);
                u32x4 o;
                o.x = pack2(f0.x, f0.y); o.y = pack2(f0.z, f0.w);
                o.z = pack2(f1.x, f1.y); o.w = pack2(f1.z, f1.w);
                ((u32x4*)dra)[j] = o;
                f0 = *(const v4f*)(gw + j * 8);
                f1 = *(const v4f*)(gw + j * 8 + 4);
                o.x = pack2(f0.x, f0.y); o.y = pack2(f0.z, f0.w);
                o.z = pack2(f1.x, f1.y); o.w = pack2(f1.z, f1.w);
                ((u32x4*)drw)[j] = o;
            }
        }
    };

    auto compute = [&](int buf) {
#pragma unroll
        for (int kc = 0; kc < BK; kc += 32) {
            Frag a[4], b[4];
#pragma unroll
            for (int i = 0; i < 4; ++i) {
                // A 16x32 bf16 frag: lane row = l15,
                // contiguous K run [kc+kbA, +15]
                const unsigned short* p =
                    &sA[buf][(wm + i * 16 + l15) * LDK + kc + kbA];
                a[i].u[0] = *(const u32x4*)p;
                a[i].u[1] = *(const u32x4*)(p + 8);
            }
#pragma unroll
            for (int j = 0; j < 4; ++j) {
                // B 32x16 bf16 frag: lane col = l15, contiguous K run
                // [kc+ksB, +15] = W[n][kc+ksB ...] row-major -> no transpose.
                const unsigned short* p =
                    &sW[buf][(wn + j * 16 + l15) * LDK + kc + ksB];
                b[j].u[0] = *(const u32x4*)p;
                b[j].u[1] = *(const u32x4*)(p + 8);
            }
#pragma unroll
            for (int i = 0; i < 4; ++i)
#pragma unroll
                for (int j = 0; j < 4; ++j)
                    acc[i][j] = __builtin_amdgcn_wmma_f32_16x16x32_bf16(
                        false, a[i].v, false, b[j].v, (short)0, acc[i][j],
                        false, false);
        }
    };

    load_tiles(0, 0);
    if constexpr (PRE) { if (HAVE_ASYNC_LDS) wait_async0(); }
    __syncthreads();
#pragma unroll 1
    for (int kt = 0; kt < NT; ++kt) {
        const int cur = kt & 1;
        if (kt + 1 < NT) load_tiles(kt + 1, cur ^ 1);
        compute(cur);
        if constexpr (PRE) { if (HAVE_ASYNC_LDS) wait_async0(); }
        __syncthreads();
    }

    // Fused epilogue: out = scale*acc + bias
    const float scl = *scalep;
#pragma unroll
    for (int j = 0; j < 4; ++j) {
        const int col = n0 + wn + j * 16 + l15;
        const float bv = bias[col];
#pragma unroll
        for (int i = 0; i < 4; ++i) {
            const int rbase = m0 + wm + i * 16 + (hi ? 8 : 0);
#pragma unroll
            for (int r = 0; r < 8; ++r)
                out[(size_t)(rbase + r) * DIM_O + col] = scl * acc[i][j][r] + bv;
        }
    }
}

// ---------------------------------------------------------------------------
extern "C" void kernel_launch(void* const* d_in, const int* in_sizes, int n_in,
                              void* d_out, int out_size, void* d_ws, size_t ws_size,
                              hipStream_t stream) {
    const float* x     = (const float*)d_in[0];   // [4096,4096]
    const float* w     = (const float*)d_in[1];   // [4096,4096]
    const float* bias  = (const float*)d_in[2];   // [4096]
    const float* scale = (const float*)d_in[3];   // [1]
    float* out = (float*)d_out;

    dim3 grid(DIM_O / BN, DIM_B / BM);            // 32 x 32 blocks

    const size_t elems = (size_t)DIM_B * DIM_I;
    const size_t need  = elems * 2 * sizeof(unsigned short);  // bf16 x + bf16 W

    if (ws_size >= need) {
        unsigned short* xb = (unsigned short*)d_ws;
        unsigned short* wb = xb + elems;
        const int cvt_blocks = (int)(elems / (256 * 8));      // 8192
        cvt_f32_bf16_kernel<<<cvt_blocks, 256, 0, stream>>>(x, xb);
        cvt_f32_bf16_kernel<<<cvt_blocks, 256, 0, stream>>>(w, wb);
        ternary_gemm_kernel<true><<<grid, 128, 0, stream>>>(
            (const void*)xb, (const void*)wb, bias, scale, out);
    } else {
        // Fallback: convert f32->bf16 inline while staging tiles to LDS.
        ternary_gemm_kernel<false><<<grid, 128, 0, stream>>>(
            (const void*)x, (const void*)w, bias, scale, out);
    }
}
